// CausalAttention_2001454760018
// MI455X (gfx1250) — compile-verified, matrix operand order checked by reading, not run on hardware
//
#include <hip/hip_runtime.h>

typedef __bf16 bf16;
typedef __attribute__((ext_vector_type(16))) __bf16 v16bf;
typedef __attribute__((ext_vector_type(8)))  __bf16 v8bf;
typedef __attribute__((ext_vector_type(8)))  float  v8f;
typedef __attribute__((ext_vector_type(4)))  int    v4i;

#define NBATCH 4
#define CDIM 256
#define SDIM 4096
#define EDIM 256
#define ODIM 256
#define NH 8
#define DHEAD 32

__device__ __forceinline__ v16bf ldg16(const bf16* p) {
  return *reinterpret_cast<const v16bf*>(p);
}
__device__ __forceinline__ v16bf lds16(const bf16* p) {
  return *reinterpret_cast<const v16bf*>(p);
}

__device__ __forceinline__ float fast_exp2(float x) {
#if defined(__HIP_DEVICE_COMPILE__) && __has_builtin(__builtin_amdgcn_exp2f)
  return __builtin_amdgcn_exp2f(x);   // v_exp_f32
#else
  return exp2f(x);
#endif
}

// ---- async global -> LDS copy of 16 bytes per lane (ASYNCcnt path) ----
__device__ __forceinline__ void async_cp16(const bf16* g, bf16* l) {
#if defined(__HIP_DEVICE_COMPILE__) && __has_builtin(__builtin_amdgcn_global_load_async_to_lds_b128)
  typedef __attribute__((address_space(1))) v4i gv4i;
  typedef __attribute__((address_space(3))) v4i sv4i;
  __builtin_amdgcn_global_load_async_to_lds_b128((gv4i*)g, (sv4i*)l, 0, 0);
#else
  *reinterpret_cast<uint4*>(l) = *reinterpret_cast<const uint4*>(g);
#endif
}
__device__ __forceinline__ void wait_async0() {
#if defined(__HIP_DEVICE_COMPILE__) && __has_builtin(__builtin_amdgcn_s_wait_asynccnt)
  __builtin_amdgcn_s_wait_asynccnt(0);
#else
  asm volatile("s_wait_asynccnt 0" ::: "memory");
#endif
}

// ---------------- conversion kernels ----------------

__global__ void cvt_w_kernel(const float* __restrict__ src, bf16* __restrict__ dst, int count) {
  int i = blockIdx.x * blockDim.x + threadIdx.x;
  if (i < count) dst[i] = (bf16)src[i];
}

// x: [n][C][S] f32  ->  xb: [n][S][C] bf16 (tiled LDS transpose)
__global__ void cvt_x_kernel(const float* __restrict__ x, bf16* __restrict__ xb) {
  __shared__ float tile[32][33];
  int n  = blockIdx.z;
  int s0 = blockIdx.x * 32, c0 = blockIdx.y * 32;
  int tx = threadIdx.x, ty = threadIdx.y;
#pragma unroll
  for (int j = 0; j < 4; ++j)
    tile[ty + 8 * j][tx] = x[(size_t)(n * CDIM + c0 + ty + 8 * j) * SDIM + s0 + tx];
  __syncthreads();
#pragma unroll
  for (int j = 0; j < 4; ++j)
    xb[(size_t)(n * SDIM + s0 + ty + 8 * j) * CDIM + c0 + tx] = (bf16)tile[tx][ty + 8 * j];
}

// ---------------- WMMA GEMM tile core ----------------
// Wave computes a 16(M) x 64(N) tile, K = 256, A reused across 4 N-subtiles.
__device__ __forceinline__ void gemm4(const bf16* __restrict__ A,
                                      const bf16* __restrict__ B,
                                      v8f acc[4], int lr, int hi) {
#pragma unroll
  for (int k0 = 0; k0 < CDIM; k0 += 32) {
    v16bf a = ldg16(A + (size_t)lr * CDIM + k0 + 16 * hi);
#pragma unroll
    for (int t = 0; t < 4; ++t) {
      v16bf b = ldg16(B + (size_t)(t * 16 + lr) * CDIM + k0 + 16 * hi);
      acc[t] = __builtin_amdgcn_wmma_f32_16x16x32_bf16(
          false, a, false, b, (short)0, acc[t], false, false);
    }
  }
}

// q = (Wq @ x + bq) * log2(e)/sqrt(dh), stored as [n][S][E] bf16
__global__ void proj_q_kernel(const bf16* __restrict__ W, const float* __restrict__ bias,
                              const bf16* __restrict__ xb, bf16* __restrict__ q) {
  int lane = threadIdx.x & 31, wave = threadIdx.x >> 5;
  int lr = lane & 15, hi = lane >> 4;
  int mbase = blockIdx.y * 128 + wave * 16;
  int sbase = blockIdx.x * 64;
  int n = blockIdx.z;
  v8f acc[4] = {};
  gemm4(W + (size_t)mbase * CDIM, xb + (size_t)(n * SDIM + sbase) * CDIM, acc, lr, hi);
  const float scale = 0.25503485f; // log2(e)/sqrt(32): softmax done in exp2 domain
#pragma unroll
  for (int t = 0; t < 4; ++t) {
    v8bf pk;
#pragma unroll
    for (int vr = 0; vr < 8; ++vr) {
      int m = mbase + hi * 8 + vr;
      pk[vr] = (bf16)((acc[t][vr] + bias[m]) * scale);
    }
    *reinterpret_cast<v8bf*>(q + (size_t)(n * SDIM + sbase + t * 16 + lr) * EDIM +
                             mbase + hi * 8) = pk;
  }
}

// kv = Wkv @ x + bkv; rows [0,256) -> k [n][S][E] bf16, rows [256,512) -> vT [n][h][d][S] bf16
__global__ void proj_kv_kernel(const bf16* __restrict__ W, const float* __restrict__ bias,
                               const bf16* __restrict__ xb, bf16* __restrict__ kOut,
                               bf16* __restrict__ vT) {
  int lane = threadIdx.x & 31, wave = threadIdx.x >> 5;
  int lr = lane & 15, hi = lane >> 4;
  int mbase = blockIdx.y * 128 + wave * 16;
  int sbase = blockIdx.x * 64;
  int n = blockIdx.z;
  v8f acc[4] = {};
  gemm4(W + (size_t)mbase * CDIM, xb + (size_t)(n * SDIM + sbase) * CDIM, acc, lr, hi);
  if (mbase < EDIM) {
#pragma unroll
    for (int t = 0; t < 4; ++t) {
      v8bf pk;
#pragma unroll
      for (int vr = 0; vr < 8; ++vr) {
        int m = mbase + hi * 8 + vr;
        pk[vr] = (bf16)(acc[t][vr] + bias[m]);
      }
      *reinterpret_cast<v8bf*>(kOut + (size_t)(n * SDIM + sbase + t * 16 + lr) * EDIM +
                               mbase + hi * 8) = pk;
    }
  } else {
#pragma unroll
    for (int t = 0; t < 4; ++t) {
#pragma unroll
      for (int vr = 0; vr < 8; ++vr) {
        int m  = mbase + hi * 8 + vr;
        int vm = m - EDIM;
        int hh = vm >> 5, dh = vm & 31;
        vT[(size_t)((n * NH + hh) * DHEAD + dh) * SDIM + sbase + t * 16 + lr] =
            (bf16)(acc[t][vr] + bias[m]);
      }
    }
  }
}

// flash attention: block = 4 waves = 64 queries; K/V tiles double-buffered in LDS
// via async global->LDS copies; 2 QK + 2 PV WMMAs per wave per 32-key block.
__global__ void attn_kernel(const bf16* __restrict__ q, const bf16* __restrict__ k,
                            const bf16* __restrict__ vT, bf16* __restrict__ ao) {
  __shared__ __align__(32) bf16 Ks[2][32 * 32];   // [key][d]
  __shared__ __align__(32) bf16 Vs[2][32 * 32];   // [d][key]
  __shared__ __align__(32) bf16 Plds[4][16 * 32];
  int tid  = threadIdx.x;
  int lane = tid & 31, wave = tid >> 5;
  int lr = lane & 15, hi = lane >> 4;
  int qb0   = blockIdx.x * 64;
  int qbase = qb0 + wave * 16;
  int head = blockIdx.y, n = blockIdx.z;
  size_t rowBase = (size_t)n * SDIM;
  int ho = head * DHEAD;

  const bf16* kg = k + rowBase * EDIM + ho;                        // key rows, stride E
  const bf16* vg = vT + (size_t)((n * NH + head) * DHEAD) * SDIM;  // d rows, stride S

  int crow = tid >> 2;        // 0..31: tile row this thread copies
  int cseg = (tid & 3) * 8;   // 16B segment within the 64B row

  v16bf aq = ldg16(q + (rowBase + qbase + lr) * EDIM + ho + 16 * hi);

  v8f oacc0 = {}, oacc1 = {};
  float mrow[8], lrow[8];
#pragma unroll
  for (int i = 0; i < 8; ++i) { mrow[i] = -1e30f; lrow[i] = 0.f; }

  int kmax = qb0 + 64;
  // prologue: fill buffer 0 with key block 0
  async_cp16(kg + (size_t)crow * EDIM + cseg, &Ks[0][crow * 32 + cseg]);
  async_cp16(vg + (size_t)crow * SDIM + cseg, &Vs[0][crow * 32 + cseg]);

  for (int kb = 0; kb < kmax; kb += 32) {
    int cur = (kb >> 5) & 1;
    wait_async0();       // own async fills done
    __syncthreads();     // everyone's fills done & prior buffer fully consumed
    if (kb + 32 < kmax) {
      int nxt = cur ^ 1;
      async_cp16(kg + (size_t)(kb + 32 + crow) * EDIM + cseg, &Ks[nxt][crow * 32 + cseg]);
      async_cp16(vg + (size_t)crow * SDIM + kb + 32 + cseg,   &Vs[nxt][crow * 32 + cseg]);
    }
    if (kb < qbase + 16) {  // causal: this wave still has live keys (uniform per wave)
      v16bf bk0 = lds16(&Ks[cur][lr * 32 + 16 * hi]);
      v16bf bk1 = lds16(&Ks[cur][(16 + lr) * 32 + 16 * hi]);
      v8f s0 = {}, s1 = {};
      s0 = __builtin_amdgcn_wmma_f32_16x16x32_bf16(false, aq, false, bk0, (short)0, s0, false, false);
      s1 = __builtin_amdgcn_wmma_f32_16x16x32_bf16(false, aq, false, bk1, (short)0, s1, false, false);

#pragma unroll
      for (int vr = 0; vr < 8; ++vr) {
        int qg = qbase + hi * 8 + vr;
        float x0 = ((kb + lr)      <= qg) ? s0[vr] : -1e30f;
        float x1 = ((kb + 16 + lr) <= qg) ? s1[vr] : -1e30f;
        float mx = fmaxf(x0, x1);
        mx = fmaxf(mx, __shfl_xor(mx, 1));
        mx = fmaxf(mx, __shfl_xor(mx, 2));
        mx = fmaxf(mx, __shfl_xor(mx, 4));
        mx = fmaxf(mx, __shfl_xor(mx, 8));
        float nm    = fmaxf(mrow[vr], mx);
        float alpha = fast_exp2(mrow[vr] - nm);   // scores already in log2 domain
        mrow[vr] = nm;
        float p0 = fast_exp2(x0 - nm);
        float p1 = fast_exp2(x1 - nm);
        float ps = p0 + p1;
        ps += __shfl_xor(ps, 1);
        ps += __shfl_xor(ps, 2);
        ps += __shfl_xor(ps, 4);
        ps += __shfl_xor(ps, 8);
        lrow[vr] = lrow[vr] * alpha + ps;
        oacc0[vr] *= alpha;
        oacc1[vr] *= alpha;
        int r = hi * 8 + vr;
        Plds[wave][r * 32 + lr]      = (bf16)p0;  // D-layout -> [q][key] tile
        Plds[wave][r * 32 + 16 + lr] = (bf16)p1;
      }
      asm volatile("s_wait_dscnt 0" ::: "memory"); // cross-lane LDS transpose handoff
      v16bf ap  = lds16(&Plds[wave][lr * 32 + 16 * hi]); // A-layout P
      v16bf bv0 = lds16(&Vs[cur][lr * 32 + 16 * hi]);        // N = d 0..15
      v16bf bv1 = lds16(&Vs[cur][(16 + lr) * 32 + 16 * hi]); // N = d 16..31
      oacc0 = __builtin_amdgcn_wmma_f32_16x16x32_bf16(false, ap, false, bv0, (short)0, oacc0, false, false);
      oacc1 = __builtin_amdgcn_wmma_f32_16x16x32_bf16(false, ap, false, bv1, (short)0, oacc1, false, false);
    }
  }

  // epilogue: normalize, transpose through LDS, coalesced bf16 store to ao[n][S][O]
  float rinv[8];
#pragma unroll
  for (int vr = 0; vr < 8; ++vr) rinv[vr] = 1.0f / lrow[vr];
#pragma unroll
  for (int vr = 0; vr < 8; ++vr) {
    int r = hi * 8 + vr;
    Plds[wave][r * 32 + lr]      = (bf16)(oacc0[vr] * rinv[vr]);
    Plds[wave][r * 32 + 16 + lr] = (bf16)(oacc1[vr] * rinv[vr]);
  }
  asm volatile("s_wait_dscnt 0" ::: "memory");
  v16bf orow = lds16(&Plds[wave][lr * 32 + 16 * hi]);
  *reinterpret_cast<v16bf*>(ao + (rowBase + qbase + lr) * ODIM + ho + 16 * hi) = orow;
}

// out = Wp @ attn + bp, f32 to d_out [n][O][S]
__global__ void proj_o_kernel(const bf16* __restrict__ W, const float* __restrict__ bias,
                              const bf16* __restrict__ ao, float* __restrict__ out) {
  int lane = threadIdx.x & 31, wave = threadIdx.x >> 5;
  int lr = lane & 15, hi = lane >> 4;
  int mbase = blockIdx.y * 128 + wave * 16;
  int sbase = blockIdx.x * 64;
  int n = blockIdx.z;
  v8f acc[4] = {};
  gemm4(W + (size_t)mbase * ODIM, ao + (size_t)(n * SDIM + sbase) * ODIM, acc, lr, hi);
#pragma unroll
  for (int t = 0; t < 4; ++t) {
#pragma unroll
    for (int vr = 0; vr < 8; ++vr) {
      int m = mbase + hi * 8 + vr;
      out[(size_t)(n * ODIM + m) * SDIM + sbase + t * 16 + lr] = acc[t][vr] + bias[m];
    }
  }
}

extern "C" void kernel_launch(void* const* d_in, const int* in_sizes, int n_in,
                              void* d_out, int out_size, void* d_ws, size_t ws_size,
                              hipStream_t stream) {
  const float* x   = (const float*)d_in[0];
  const float* Wq  = (const float*)d_in[1];
  const float* bq  = (const float*)d_in[2];
  const float* Wkv = (const float*)d_in[3];
  const float* bkv = (const float*)d_in[4];
  const float* Wp  = (const float*)d_in[5];
  const float* bp  = (const float*)d_in[6];
  float* out = (float*)d_out;

  char* ws = (char*)d_ws;
  size_t off = 0;
  auto carve = [&](size_t bytes) {
    char* p = ws + off;
    off += (bytes + 255) & ~(size_t)255;
    return p;
  };
  bf16* xb   = (bf16*)carve((size_t)NBATCH * SDIM * CDIM * 2);
  bf16* qb   = (bf16*)carve((size_t)NBATCH * SDIM * EDIM * 2);
  bf16* kbuf = (bf16*)carve((size_t)NBATCH * SDIM * EDIM * 2);
  bf16* vt   = (bf16*)carve((size_t)NBATCH * NH * DHEAD * SDIM * 2);
  bf16* ao   = (bf16*)carve((size_t)NBATCH * SDIM * ODIM * 2);
  bf16* Wqb  = (bf16*)carve((size_t)EDIM * CDIM * 2);
  bf16* Wkvb = (bf16*)carve((size_t)(EDIM + ODIM) * CDIM * 2);
  bf16* Wpb  = (bf16*)carve((size_t)ODIM * ODIM * 2);

  cvt_w_kernel<<<(EDIM * CDIM + 255) / 256, 256, 0, stream>>>(Wq, Wqb, EDIM * CDIM);
  cvt_w_kernel<<<((EDIM + ODIM) * CDIM + 255) / 256, 256, 0, stream>>>(Wkv, Wkvb, (EDIM + ODIM) * CDIM);
  cvt_w_kernel<<<(ODIM * ODIM + 255) / 256, 256, 0, stream>>>(Wp, Wpb, ODIM * ODIM);
  cvt_x_kernel<<<dim3(SDIM / 32, CDIM / 32, NBATCH), dim3(32, 8), 0, stream>>>(x, xb);

  proj_q_kernel<<<dim3(SDIM / 64, EDIM / 128, NBATCH), 256, 0, stream>>>(Wqb, bq, xb, qb);
  proj_kv_kernel<<<dim3(SDIM / 64, (EDIM + ODIM) / 128, NBATCH), 256, 0, stream>>>(Wkvb, bkv, xb, kbuf, vt);
  attn_kernel<<<dim3(SDIM / 64, NH, NBATCH), 128, 0, stream>>>(qb, kbuf, vt, ao);
  proj_o_kernel<<<dim3(SDIM / 64, ODIM / 128, NBATCH), 256, 0, stream>>>(Wpb, bp, ao, out);
}
